// SeqTokenAlignerOT_6322191859875
// MI455X (gfx1250) — matrix operand order repaired and checked by reading, not compiled, not run on hardware
//
#include <hip/hip_runtime.h>
#include <hip/hip_bf16.h>

// ---------------------------------------------------------------------------
// Problem constants (match reference)
// ---------------------------------------------------------------------------
#define S_SEQ 2048
#define D_DIM 1024
#define REG_INV 10.0f              // 1/0.1
#define FI_EXP 0.0099009900990099f // reg_m/(reg_m+reg) = 0.001/0.101
#define N_ITERS 250

// ---------------------------------------------------------------------------
// WMMA vector types (gfx1250, wave32)
// ---------------------------------------------------------------------------
typedef __attribute__((ext_vector_type(16))) __bf16 bf16x16;
typedef __attribute__((ext_vector_type(8)))  float  f32x8;

struct FragU { uint4 lo, hi; };   // 32B raw fragment -> bit_cast to bf16x16

// f32 -> bf16 bits, round-to-nearest-even
__device__ __forceinline__ unsigned short f2bf(float x) {
  union { float f; unsigned int u; } c; c.f = x;
  unsigned int u = c.u;
  unsigned int r = (u + 0x7FFFu + ((u >> 16) & 1u)) >> 16;
  return (unsigned short)r;
}

// ---------------------------------------------------------------------------
// Elementwise: f32 -> bf16 convert
// ---------------------------------------------------------------------------
__global__ __launch_bounds__(256) void f32_to_bf16_kernel(
    const float* __restrict__ in, unsigned short* __restrict__ out, int n) {
  int i = blockIdx.x * 256 + threadIdx.x;
  if (i < n) out[i] = f2bf(in[i]);
}

// ---------------------------------------------------------------------------
// Row L2-normalize (f32 in) -> bf16 out. One block per row, D = 1024.
// ---------------------------------------------------------------------------
__global__ __launch_bounds__(256) void rownorm_bf16_kernel(
    const float* __restrict__ src, unsigned short* __restrict__ dst, int D) {
  __shared__ float red[8];
  int row = blockIdx.x;
  const float* r = src + (size_t)row * D;
  float s = 0.f;
  for (int t = threadIdx.x; t < D; t += 256) { float x = r[t]; s += x * x; }
  for (int off = 16; off; off >>= 1) s += __shfl_down(s, off, 32);
  int lane = threadIdx.x & 31, wave = threadIdx.x >> 5;
  if (lane == 0) red[wave] = s;
  __syncthreads();
  if (threadIdx.x == 0) {
    float t = 0.f;
    #pragma unroll
    for (int i = 0; i < 8; ++i) t += red[i];
    red[0] = 1.f / fmaxf(sqrtf(t), 1e-8f);
  }
  __syncthreads();
  float inv = red[0];
  unsigned short* o = dst + (size_t)row * D;
  for (int t = threadIdx.x; t < D; t += 256) o[t] = f2bf(r[t] * inv);
}

// ---------------------------------------------------------------------------
// K = exp(A / reg); also store K^T (so both Sinkhorn matvecs stream rows)
// ---------------------------------------------------------------------------
__global__ __launch_bounds__(256) void expk_kernel(
    const float* __restrict__ Asim, float* __restrict__ Km,
    float* __restrict__ KmT) {
  int idx = blockIdx.x * 256 + threadIdx.x;
  int i = idx >> 11, j = idx & (S_SEQ - 1);
  float v = expf(Asim[idx] * REG_INV);
  Km[idx] = v;
  KmT[(size_t)j * S_SEQ + i] = v;
}

__global__ __launch_bounds__(256) void init_uv_kernel(
    float* __restrict__ u, float* __restrict__ v) {
  int i = blockIdx.x * 256 + threadIdx.x;
  if (i < S_SEQ) { u[i] = 1.0f / S_SEQ; v[i] = 1.0f / S_SEQ; }
}

// ---------------------------------------------------------------------------
// One Sinkhorn half-step: y[i] = ( (1/S) / dot(M[i,:], x) )^fi
// One wave per row; float4 streaming (L2-resident M).
// ---------------------------------------------------------------------------
__global__ __launch_bounds__(256) void sinkhorn_step_kernel(
    const float* __restrict__ M, const float* __restrict__ x,
    float* __restrict__ y) {
  int wave = threadIdx.x >> 5, lane = threadIdx.x & 31;
  int row = blockIdx.x * 8 + wave;
  const float4* Mr = (const float4*)(M + (size_t)row * S_SEQ);
  const float4* x4 = (const float4*)x;
  float s = 0.f;
  #pragma unroll 4
  for (int t = lane; t < (S_SEQ / 4); t += 32) {
    float4 m = Mr[t], xx = x4[t];
    s += m.x * xx.x + m.y * xx.y + m.z * xx.z + m.w * xx.w;
  }
  for (int off = 16; off; off >>= 1) s += __shfl_down(s, off, 32);
  if (lane == 0) y[row] = powf((1.0f / S_SEQ) / s, FI_EXP);
}

// ---------------------------------------------------------------------------
// T = A * (u K v^T) + delta, emitted as bf16
// ---------------------------------------------------------------------------
__global__ __launch_bounds__(256) void build_T_kernel(
    const float* __restrict__ Asim, const float* __restrict__ Km,
    const float* __restrict__ u, const float* __restrict__ v,
    const float* __restrict__ delta, unsigned short* __restrict__ T) {
  int idx = blockIdx.x * 256 + threadIdx.x;
  int i = idx >> 11, j = idx & (S_SEQ - 1);
  float w = u[i] * Km[idx] * v[j];
  T[idx] = f2bf(Asim[idx] * w + delta[idx]);
}

// ---------------------------------------------------------------------------
// XaT[n, k] = bf16(Xal[k, n]);  Xal: [S, D], XaT: [D, S]
// ---------------------------------------------------------------------------
__global__ __launch_bounds__(256) void transpose_bf16_kernel(
    const float* __restrict__ src, unsigned short* __restrict__ dst) {
  int idx = blockIdx.x * 256 + threadIdx.x;     // over D*S, k fast
  int n = idx >> 11, k = idx & (S_SEQ - 1);
  dst[idx] = f2bf(src[(size_t)k * D_DIM + n]);
}

// ---------------------------------------------------------------------------
// bf16 WMMA GEMM, NT:  C[M,N](f32) = A[M,K](bf16) * B[N,K](bf16)^T  (+ bias)
// Block tile 64x128, K-step 32; 8 waves, each wave -> 32x32 C via 2x2 WMMA.
// Register-staged pipeline: next global tile fetched before this tile's MACs.
// Fragments loaded as ds_load_b128 pairs (layouts are K-contiguous in LDS).
// Requires M%64==0, N%128==0, K%32==0.
// ---------------------------------------------------------------------------
#define BM 64
#define BN 128
#define BK 32
#define LDSP 40   // LDS pitch in bf16 elems: 80B rows -> 16B aligned, conflict-free

__global__ __launch_bounds__(256) void gemm_bf16_nt_kernel(
    const unsigned short* __restrict__ A,  // [M,K]
    const unsigned short* __restrict__ B,  // [N,K]
    float* __restrict__ C,                 // [M,N]
    const float* __restrict__ bias,        // [N] or nullptr
    int M, int N, int K) {
  __shared__ __align__(16) unsigned short lA[BM * LDSP];
  __shared__ __align__(16) unsigned short lB[BN * LDSP];

  const int tid  = threadIdx.x;
  const int lane = tid & 31;
  const int wave = tid >> 5;     // 0..7
  const int wm   = wave >> 2;    // 0..1
  const int wn   = wave & 3;     // 0..3
  const int m0   = blockIdx.y * BM;
  const int n0   = blockIdx.x * BN;

  const int arow = tid >> 2;           // 0..63
  const int aseg = (tid & 3) * 8;      // 0,8,16,24 (bf16 elems -> 16B)

  f32x8 c00 = {}, c01 = {}, c10 = {}, c11 = {};

  const int mlane = lane & 15;
  const int hi    = lane >> 4;
  const int am0   = wm * 32 + mlane;   // local A rows for the two A fragments
  const int am1   = am0 + 16;
  const int bn0   = wn * 32 + mlane;   // local B rows for the two B fragments
  const int bn1   = bn0 + 16;
  const int kbA   = hi * 8;            // bf16-element K base (A halves)
  const int kbB   = hi * 8;            // bf16-element K base (B halves)

  // global streaming pointers (element offsets; 16B aligned)
  const unsigned short* gA  = A + (size_t)(m0 + arow) * K + aseg;
  const unsigned short* gB0 = B + (size_t)(n0 + arow) * K + aseg;
  const unsigned short* gB1 = B + (size_t)(n0 + arow + 64) * K + aseg;

  // prologue: fetch tile 0 into registers
  uint4 sa  = *(const uint4*)gA;
  uint4 sb0 = *(const uint4*)gB0;
  uint4 sb1 = *(const uint4*)gB1;

  for (int k0 = 0; k0 < K; k0 += BK) {
    __syncthreads();   // previous tile's readers done
    *(uint4*)&lA[arow * LDSP + aseg]        = sa;
    *(uint4*)&lB[arow * LDSP + aseg]        = sb0;
    *(uint4*)&lB[(arow + 64) * LDSP + aseg] = sb1;
    __syncthreads();   // tile visible

    // issue next tile's global loads before this tile's MACs (latency overlap)
    if (k0 + BK < K) {
      sa  = *(const uint4*)(gA  + k0 + BK);
      sb0 = *(const uint4*)(gB0 + k0 + BK);
      sb1 = *(const uint4*)(gB1 + k0 + BK);
      __builtin_prefetch(gA + k0 + 2 * BK, 0, 1);
      __builtin_prefetch(gB0 + k0 + 2 * BK, 0, 1);
    }

    // --- fragment gathers: 2x b128 per fragment (K-contiguous runs) ---
    FragU fa0, fa1, fb0, fb1;
    fa0.lo = *(const uint4*)&lA[am0 * LDSP + kbA];
    fa0.hi = *(const uint4*)&lA[am0 * LDSP + kbA + 16];
    fa1.lo = *(const uint4*)&lA[am1 * LDSP + kbA];
    fa1.hi = *(const uint4*)&lA[am1 * LDSP + kbA + 16];
    fb0.lo = *(const uint4*)&lB[bn0 * LDSP + kbB];
    fb0.hi = *(const uint4*)&lB[bn0 * LDSP + kbB + 16];
    fb1.lo = *(const uint4*)&lB[bn1 * LDSP + kbB];
    fb1.hi = *(const uint4*)&lB[bn1 * LDSP + kbB + 16];

    bf16x16 a0 = __builtin_bit_cast(bf16x16, fa0);
    bf16x16 a1 = __builtin_bit_cast(bf16x16, fa1);
    bf16x16 b0 = __builtin_bit_cast(bf16x16, fb0);
    bf16x16 b1 = __builtin_bit_cast(bf16x16, fb1);

    c00 = __builtin_amdgcn_wmma_f32_16x16x32_bf16(false, a0, false, b0,
                                                  (short)0, c00, false, false);
    c01 = __builtin_amdgcn_wmma_f32_16x16x32_bf16(false, a0, false, b1,
                                                  (short)0, c01, false, false);
    c10 = __builtin_amdgcn_wmma_f32_16x16x32_bf16(false, a1, false, b0,
                                                  (short)0, c10, false, false);
    c11 = __builtin_amdgcn_wmma_f32_16x16x32_bf16(false, a1, false, b1,
                                                  (short)0, c11, false, false);
  }

  // --- epilogue: C layout (ISA): row = r + 8*hi, col = lane&15 ---
  const int row0 = m0 + wm * 32 + 8 * hi;
  const int col0 = n0 + wn * 32 + mlane;
  float bb0 = bias ? bias[col0]      : 0.f;
  float bb1 = bias ? bias[col0 + 16] : 0.f;
  #pragma unroll
  for (int r = 0; r < 8; ++r) {
    int rr = row0 + r;
    C[(size_t)rr * N + col0]             = c00[r] + bb0;
    C[(size_t)rr * N + col0 + 16]        = c01[r] + bb1;
    C[(size_t)(rr + 16) * N + col0]      = c10[r] + bb0;
    C[(size_t)(rr + 16) * N + col0 + 16] = c11[r] + bb1;
  }
}

// ---------------------------------------------------------------------------
// Host-side orchestration
// ---------------------------------------------------------------------------
extern "C" void kernel_launch(void* const* d_in, const int* in_sizes, int n_in,
                              void* d_out, int out_size, void* d_ws, size_t ws_size,
                              hipStream_t stream) {
  (void)in_sizes; (void)n_in; (void)out_size; (void)ws_size;
  const float* X     = (const float*)d_in[0];  // [S, D]
  const float* Y     = (const float*)d_in[1];  // [S, D]
  const float* W_lin = (const float*)d_in[2];  // [D, D]
  const float* b_lin = (const float*)d_in[3];  // [D]
  const float* delta = (const float*)d_in[4];  // [S, S]
  float* out = (float*)d_out;                  // [S, D]

  const int S = S_SEQ, D = D_DIM;

  char* base = (char*)d_ws;
  size_t off = 0;
  auto alloc = [&](size_t bytes) -> void* {
    void* p = base + off;
    off += (bytes + 255) & ~(size_t)255;
    return p;
  };
  unsigned short* Xbf  = (unsigned short*)alloc((size_t)S * D * 2);
  unsigned short* Wbf  = (unsigned short*)alloc((size_t)D * D * 2);
  float*          Xal  = (float*)         alloc((size_t)S * D * 4);
  unsigned short* An   = (unsigned short*)alloc((size_t)S * D * 2);
  unsigned short* Bn   = (unsigned short*)alloc((size_t)S * D * 2);
  float*          Asim = (float*)         alloc((size_t)S * S * 4);
  float*          Km   = (float*)         alloc((size_t)S * S * 4);
  float*          KmT  = (float*)         alloc((size_t)S * S * 4);
  float*          u    = (float*)         alloc((size_t)S * 4);
  float*          v    = (float*)         alloc((size_t)S * 4);
  unsigned short* Tbf  = (unsigned short*)alloc((size_t)S * S * 2);
  unsigned short* XaT  = (unsigned short*)alloc((size_t)D * S * 2);

  // 1) bf16 conversions of X and W_lin
  f32_to_bf16_kernel<<<(S * D) / 256, 256, 0, stream>>>(X, Xbf, S * D);
  f32_to_bf16_kernel<<<(D * D) / 256, 256, 0, stream>>>(W_lin, Wbf, D * D);

  // 2) X_aligned = X @ W_lin^T + b   (NT: B = W_lin [N=D, K=D])
  gemm_bf16_nt_kernel<<<dim3(D / BN, S / BM), 256, 0, stream>>>(
      Xbf, Wbf, Xal, b_lin, S, D, D);

  // 3) row-normalize both operands -> bf16
  rownorm_bf16_kernel<<<S, 256, 0, stream>>>(Xal, An, D);
  rownorm_bf16_kernel<<<S, 256, 0, stream>>>(Y, Bn, D);

  // 4) A = An @ Bn^T  (cosine similarity)
  gemm_bf16_nt_kernel<<<dim3(S / BN, S / BM), 256, 0, stream>>>(
      An, Bn, Asim, nullptr, S, S, D);

  // 5) K = exp(A/reg) and K^T; init u, v
  expk_kernel<<<(S * S) / 256, 256, 0, stream>>>(Asim, Km, KmT);
  init_uv_kernel<<<(S + 255) / 256, 256, 0, stream>>>(u, v);

  // 6) 250 unbalanced Sinkhorn iterations (L2-resident matvecs)
  for (int it = 0; it < N_ITERS; ++it) {
    sinkhorn_step_kernel<<<S / 8, 256, 0, stream>>>(Km, v, u);
    sinkhorn_step_kernel<<<S / 8, 256, 0, stream>>>(KmT, u, v);
  }

  // 7) T = A * (u K v^T) + delta  (bf16), and X_aligned^T (bf16)
  build_T_kernel<<<(S * S) / 256, 256, 0, stream>>>(Asim, Km, u, v, delta, Tbf);
  transpose_bf16_kernel<<<(D * S) / 256, 256, 0, stream>>>(Xal, XaT);

  // 8) out = T @ X_aligned   (NT with B = XaT [N=D, K=S])
  gemm_bf16_nt_kernel<<<dim3(D / BN, S / BM), 256, 0, stream>>>(
      Tbf, XaT, out, nullptr, S, D, S);
}
